// GNN_module_34299608826047
// MI455X (gfx1250) — matrix operand order repaired and checked by reading, not compiled, not run on hardware
//
#include <hip/hip_runtime.h>
#include <math.h>

// ---------------- WMMA types / helpers (CDNA5 gfx1250, wave32) ----------------
typedef float   v8f   __attribute__((ext_vector_type(8)));
typedef __bf16  v16bf __attribute__((ext_vector_type(16)));

union V16 { v16bf v; uint4 q[2]; };   // 32 bytes: one WMMA bf16 operand per lane

__device__ __forceinline__ v8f wmma_bf16(v16bf a, v16bf b, v8f c) {
    // D = A(16x32 bf16) * B(32x16 bf16) + C(16x16 f32)
    return __builtin_amdgcn_wmma_f32_16x16x32_bf16(false, a, false, b, (short)0, c, false, false);
}

__device__ __forceinline__ unsigned short f2bf(float f) {   // RNE f32 -> bf16
    unsigned int u = __float_as_uint(f);
    u += 0x7FFFu + ((u >> 16) & 1u);
    return (unsigned short)(u >> 16);
}
__device__ __forceinline__ unsigned int pack_bf2(float a, float b) {
    return (unsigned int)f2bf(a) | ((unsigned int)f2bf(b) << 16);
}

// ---------------- problem constants ----------------
#define B_  32
#define V_  96
#define XS  448          // feature buffer stride (max cin)
#define C1  256          // adj conv stage1 out
#define C2  128          // adj conv stage2 out
#define NE_ADJ  (B_*V_*V_)   // 294912 edges (BN population for adj convs)
#define NE_G    (B_*V_)      // 3072 rows (BN population for gconv)

// LDS layout for k_adj (dynamic shared, 38016 bytes)
#define S_X      0                  // 448 f32
#define S_PHI    1792               // 96 x 40 u16 (32 data + 8 pad; 80B rows -> conflict-free b128)
#define S_W      9472               // 256 x 40 u16 (weight chunk, transposed [N][K])
#define S_CHUNK  29952              // 96 x 40 u16 (stage2 A-operand staging)
#define S_LOGIT  37632              // 96 f32
#define SMEM_ADJ 38016
#define PHI_ST   40                 // u16 row stride

// ============================================================================
// Weight prep: Wbf[n*K + k] = bf16(W[k*N + n])  (transpose + convert, once/layer)
// ============================================================================
__global__ void k_wprep(const float* __restrict__ W, unsigned short* __restrict__ Wbf,
                        int K, int N) {
    int idx = blockIdx.x * blockDim.x + threadIdx.x;
    if (idx < K * N) {
        int n = idx / K, k = idx - n * K;
        Wbf[idx] = f2bf(W[(size_t)k * N + n]);
    }
}

// ============================================================================
// Fused adjacency-MLP kernel. One block per (b,i): 96 edges x full MLP.
// STAGE 1: GEMM1 (phi @ W1), accumulate BN1 stats.
// STAGE 2: GEMM1 -> BN1+leaky -> GEMM2, accumulate BN2 stats.
// STAGE 3: GEMM1 -> BN1 -> GEMM2 -> BN2+leaky -> dot(w3) -> softmax_j -> A.
// 8 waves: wave w owns stage1 N-tiles {2w,2w+1} == stage2 K-chunk [32w,32w+32).
// ============================================================================
template <int STAGE>
__global__ __launch_bounds__(256)
void k_adj(const float* __restrict__ xbuf, int cin,
           const unsigned short* __restrict__ W1bf,   // [C1][cin] bf16 (transposed)
           const float* __restrict__ b1,
           const float* __restrict__ sc1, const float* __restrict__ sh1,
           const unsigned short* __restrict__ W2bf,   // [C2][C1] bf16 (transposed)
           const float* __restrict__ b2,
           const float* __restrict__ sc2, const float* __restrict__ sh2,
           const float* __restrict__ w3, const float* __restrict__ b3,
           float* __restrict__ sum1, float* __restrict__ sq1,
           float* __restrict__ sum2, float* __restrict__ sq2,
           float* __restrict__ Aout)
{
    extern __shared__ __align__(16) char smem[];
    float*          sX  = (float*)(smem + S_X);
    unsigned short* sPh = (unsigned short*)(smem + S_PHI);
    unsigned short* sW  = (unsigned short*)(smem + S_W);
    unsigned short* sCk = (unsigned short*)(smem + S_CHUNK);
    float*          sLg = (float*)(smem + S_LOGIT);

    const int bi   = blockIdx.x;            // 0 .. B*V-1
    const int b    = bi / V_;
    const int i    = bi % V_;
    const int tid  = threadIdx.x;
    const int wave = tid >> 5;
    const int lane = tid & 31;
    const int half = lane >> 4;             // 0/1: which 16-lane half
    const int ln   = lane & 15;             // N index within tile / M row within tile

    const float* xb = xbuf + (size_t)b * V_ * XS;
    const float* xi = xb + (size_t)i * XS;

    for (int c = tid; c < cin; c += 256) sX[c] = xi[c];
    if (STAGE == 3) { if (tid < V_) sLg[tid] = b3[0]; }

    // -------- stage 1: acc1[p][mt] = phi(96 x cin) @ W1 tile --------
    v8f acc1[2][6];
    #pragma unroll
    for (int p = 0; p < 2; ++p)
        #pragma unroll
        for (int mt = 0; mt < 6; ++mt)
            #pragma unroll
            for (int r = 0; r < 8; ++r) acc1[p][mt][r] = 0.0f;

    const int KC = cin >> 5;
    for (int kc = 0; kc < KC; ++kc) {
        const int k0 = kc << 5;
        __syncthreads();
        // phi chunk (vectorized): 96 rows x 8 groups of 4 channels
        for (int t = tid; t < V_ * 8; t += 256) {
            int j = t >> 3, g = (t & 7) << 2;
            const float4 xv  = *(const float4*)(xb + (size_t)j * XS + k0 + g);
            const float4 xiv = *(const float4*)(sX + k0 + g);
            uint2 pk;
            pk.x = pack_bf2(fabsf(xiv.x - xv.x), fabsf(xiv.y - xv.y));
            pk.y = pack_bf2(fabsf(xiv.z - xv.z), fabsf(xiv.w - xv.w));
            *(uint2*)(sPh + j * PHI_ST + g) = pk;
        }
        // W1 chunk: straight b128 copies of pre-transposed bf16 weights
        #pragma unroll
        for (int u = tid; u < C1 * 4; u += 256) {       // 1024 uint4, 4/thread
            int n = u >> 2, seg = (u & 3) << 3;
            *(uint4*)(sW + n * PHI_ST + seg) =
                *(const uint4*)(W1bf + (size_t)n * cin + k0 + seg);
        }
        __syncthreads();

        V16 bop0, bop1;
        {   // B operand: lane ln holds column N, K = half*16 .. half*16+15
            const unsigned short* bp = sW + ((wave * 2 + 0) * 16 + ln) * PHI_ST + half * 16;
            bop0.q[0] = *(const uint4*)(bp);  bop0.q[1] = *(const uint4*)(bp + 8);
            bp = sW + ((wave * 2 + 1) * 16 + ln) * PHI_ST + half * 16;
            bop1.q[0] = *(const uint4*)(bp);  bop1.q[1] = *(const uint4*)(bp + 8);
        }
        #pragma unroll
        for (int mt = 0; mt < 6; ++mt) {
            V16 aop;  // A operand: lane ln = row M, K = half*16 .. half*16+15
            const unsigned short* ap = sPh + (mt * 16 + ln) * PHI_ST + half * 16;
            aop.q[0] = *(const uint4*)(ap);
            aop.q[1] = *(const uint4*)(ap + 8);
            acc1[0][mt] = wmma_bf16(aop.v, bop0.v, acc1[0][mt]);
            acc1[1][mt] = wmma_bf16(aop.v, bop1.v, acc1[1][mt]);
        }
    }

    // bias + (stats | BN1+leaky)
    {
        float bias[2] = { b1[wave * 32 + ln], b1[wave * 32 + 16 + ln] };
        if (STAGE == 1) {
            #pragma unroll
            for (int p = 0; p < 2; ++p) {
                float s = 0.f, q = 0.f;
                #pragma unroll
                for (int mt = 0; mt < 6; ++mt)
                    #pragma unroll
                    for (int r = 0; r < 8; ++r) {
                        float y = acc1[p][mt][r] + bias[p];
                        s += y; q += y * y;
                    }
                s += __shfl_xor(s, 16, 32);
                q += __shfl_xor(q, 16, 32);
                if (half == 0) {
                    atomicAdd(&sum1[wave * 32 + p * 16 + ln], s);
                    atomicAdd(&sq1 [wave * 32 + p * 16 + ln], q);
                }
            }
            return;
        }
        #pragma unroll
        for (int p = 0; p < 2; ++p) {
            int n = wave * 32 + p * 16 + ln;
            float sc = sc1[n], sh = sh1[n];
            #pragma unroll
            for (int mt = 0; mt < 6; ++mt)
                #pragma unroll
                for (int r = 0; r < 8; ++r) {
                    float y = (acc1[p][mt][r] + bias[p]) * sc + sh;
                    acc1[p][mt][r] = (y > 0.f) ? y : 0.01f * y;     // leaky 0.01
                }
        }
    }

    // -------- stage 2: act1(96 x 256) @ W2 ; wave w owns N-tile w --------
    v8f acc2[6];
    #pragma unroll
    for (int mt = 0; mt < 6; ++mt)
        #pragma unroll
        for (int r = 0; r < 8; ++r) acc2[mt][r] = 0.0f;

    for (int kc2 = 0; kc2 < C1 / 32; ++kc2) {            // 8 chunks == 8 waves
        __syncthreads();
        #pragma unroll
        for (int u = tid; u < C2 * 4; u += 256) {        // W2 chunk: 512 uint4, 2/thread
            int n = u >> 2, seg = (u & 3) << 3;
            *(uint4*)(sW + n * PHI_ST + seg) =
                *(const uint4*)(W2bf + (size_t)n * C1 + kc2 * 32 + seg);
        }
        if (wave == kc2) {                               // this wave's act1 K-chunk
            #pragma unroll
            for (int p = 0; p < 2; ++p)
                #pragma unroll
                for (int mt = 0; mt < 6; ++mt)
                    #pragma unroll
                    for (int r = 0; r < 8; ++r) {
                        int m = mt * 16 + half * 8 + r;
                        sCk[m * PHI_ST + p * 16 + ln] = f2bf(acc1[p][mt][r]);
                    }
        }
        __syncthreads();

        V16 bop;
        const unsigned short* bp = sW + (wave * 16 + ln) * PHI_ST + half * 16;
        bop.q[0] = *(const uint4*)(bp);  bop.q[1] = *(const uint4*)(bp + 8);
        #pragma unroll
        for (int mt = 0; mt < 6; ++mt) {
            V16 aop;
            const unsigned short* ap = sCk + (mt * 16 + ln) * PHI_ST + half * 16;
            aop.q[0] = *(const uint4*)(ap);
            aop.q[1] = *(const uint4*)(ap + 8);
            acc2[mt] = wmma_bf16(aop.v, bop.v, acc2[mt]);
        }
    }

    const int   n2    = wave * 16 + ln;
    const float bias2 = b2[n2];

    if (STAGE == 2) {
        float s = 0.f, q = 0.f;
        #pragma unroll
        for (int mt = 0; mt < 6; ++mt)
            #pragma unroll
            for (int r = 0; r < 8; ++r) {
                float y = acc2[mt][r] + bias2;
                s += y; q += y * y;
            }
        s += __shfl_xor(s, 16, 32);
        q += __shfl_xor(q, 16, 32);
        if (half == 0) { atomicAdd(&sum2[n2], s); atomicAdd(&sq2[n2], q); }
        return;
    }

    // -------- stage 3: BN2 + leaky, dot with w3, softmax over j --------
    {
        float sc = sc2[n2], sh = sh2[n2], wv = w3[n2];
        __syncthreads();
        #pragma unroll
        for (int mt = 0; mt < 6; ++mt)
            #pragma unroll
            for (int r = 0; r < 8; ++r) {
                float y = (acc2[mt][r] + bias2) * sc + sh;
                y = (y > 0.f) ? y : 0.01f * y;
                float t = y * wv;
                t += __shfl_xor(t, 1, 32);
                t += __shfl_xor(t, 2, 32);
                t += __shfl_xor(t, 4, 32);
                t += __shfl_xor(t, 8, 32);               // sum over 16 channels of this half
                if (ln == 0) atomicAdd(&sLg[mt * 16 + half * 8 + r], t);  // ds_add_f32
            }
        __syncthreads();
        if (tid == 0) {                                  // softmax over 96 neighbors
            float mx = -1e30f;
            for (int j = 0; j < V_; ++j) mx = fmaxf(mx, sLg[j]);
            float sm = 0.f;
            for (int j = 0; j < V_; ++j) { float e = __expf(sLg[j] - mx); sLg[j] = e; sm += e; }
            float inv = 1.0f / sm;
            for (int j = 0; j < V_; ++j) sLg[j] *= inv;
        }
        __syncthreads();
        if (tid < V_) Aout[(size_t)bi * V_ + tid] = sLg[tid];
    }
}

// ---------------- small support kernels ----------------
__global__ void k_zero(float* p, int n) {
    int i = blockIdx.x * blockDim.x + threadIdx.x;
    if (i < n) p[i] = 0.0f;
}

__global__ void k_init_x(const float* __restrict__ xin, float* __restrict__ xbuf) {
    int bi = blockIdx.x;
    xbuf[(size_t)bi * XS + threadIdx.x] = xin[(size_t)bi * 256 + threadIdx.x];
}

// scale = g * rsqrt(var+eps); shift = be - mean*scale   (biased var, torch-style)
__global__ void k_finalize(const float* sum, const float* sq,
                           const float* g, const float* be,
                           float* sc, float* sh, int C, float invN) {
    int c = blockIdx.x * blockDim.x + threadIdx.x;
    if (c < C) {
        float m = sum[c] * invN;
        float v = sq[c] * invN - m * m;
        float s = g[c] * rsqrtf(v + 1e-5f);
        sc[c] = s;
        sh[c] = be[c] - m * s;
    }
}

// z[b,i,:] = (A[b,i,:] @ x[b]) @ Wg + bg ; accumulate BN stats
__global__ __launch_bounds__(256)
void k_gconv_z(const float* __restrict__ xbuf, const float* __restrict__ A,
               const float* __restrict__ Wg, const float* __restrict__ bg,
               float* __restrict__ z, float* sumg, float* sqg, int cin)
{
    __shared__ float sA[V_];
    __shared__ float sAgg[XS];
    __shared__ float sPart[256];
    int bi = blockIdx.x, tid = threadIdx.x;
    int b = bi / V_;
    if (tid < V_) sA[tid] = A[(size_t)bi * V_ + tid];
    __syncthreads();
    const float* xb = xbuf + (size_t)b * V_ * XS;
    for (int c = tid; c < cin; c += 256) {
        float acc = 0.f;
        for (int j = 0; j < V_; ++j) acc += sA[j] * xb[(size_t)j * XS + c];
        sAgg[c] = acc;
    }
    __syncthreads();
    int n = tid & 63, part = tid >> 6;                   // 4-way split of c
    float acc = 0.f;
    for (int c = part; c < cin; c += 4) acc += sAgg[c] * Wg[(size_t)c * 64 + n];
    sPart[tid] = acc;
    __syncthreads();
    if (tid < 64) {
        float v = sPart[tid] + sPart[tid + 64] + sPart[tid + 128] + sPart[tid + 192] + bg[tid];
        z[(size_t)bi * 64 + tid] = v;
        atomicAdd(&sumg[tid], v);
        atomicAdd(&sqg[tid], v * v);
    }
}

// x[b,i,cin+n] = leaky0.1(BN(z))
__global__ void k_gconv_apply(const float* __restrict__ z,
                              const float* __restrict__ scg, const float* __restrict__ shg,
                              float* __restrict__ xbuf, int cin)
{
    int idx = blockIdx.x * blockDim.x + threadIdx.x;     // B*V*64 total
    int bi = idx >> 6, n = idx & 63;
    float y = z[idx] * scg[n] + shg[n];
    y = (y > 0.f) ? y : 0.1f * y;
    xbuf[(size_t)bi * XS + cin + n] = y;
}

// out[b,:] = (A_last[b,0,:] @ x[b]) @ Wl + bl   (only row i=0 is needed)
__global__ __launch_bounds__(256)
void k_last_out(const float* __restrict__ xbuf, const float* __restrict__ Alast,
                const float* __restrict__ Wl, const float* __restrict__ bl,
                float* __restrict__ out)
{
    __shared__ float sA[V_];
    __shared__ float sAgg[XS];
    int b = blockIdx.x, tid = threadIdx.x;
    if (tid < V_) sA[tid] = Alast[(size_t)b * V_ * V_ + tid];
    __syncthreads();
    const float* xb = xbuf + (size_t)b * V_ * XS;
    for (int c = tid; c < XS; c += 256) {
        float a = 0.f;
        for (int j = 0; j < V_; ++j) a += sA[j] * xb[(size_t)j * XS + c];
        sAgg[c] = a;
    }
    __syncthreads();
    if (tid < 5) {
        float a = bl[tid];
        for (int c = 0; c < XS; ++c) a += sAgg[c] * Wl[c * 5 + tid];
        out[b * 5 + tid] = a;
    }
}

// ============================================================================
// host launcher
// ============================================================================
extern "C" void kernel_launch(void* const* d_in, const int* in_sizes, int n_in,
                              void* d_out, int out_size, void* d_ws, size_t ws_size,
                              hipStream_t stream)
{
    (void)in_sizes; (void)n_in; (void)out_size; (void)ws_size;

    // d_in flattened in setup_inputs() insertion order:
    // 0:x | per layer: adj{c0:w,b,g,be | c1:w,b,g,be | c2:w,b} gconv{w,b,g,be}
    // then last_adj (same 10), last_conv {w,b}
    struct AdjIdx { int w1, b1, g1, be1, w2, b2, g2, be2, w3, b3; };
    const AdjIdx AJ[4] = {
        { 1, 2, 3, 4, 5, 6, 7, 8, 9,10},
        {15,16,17,18,19,20,21,22,23,24},
        {29,30,31,32,33,34,35,36,37,38},
        {43,44,45,46,47,48,49,50,51,52}};
    const int GW[3][4] = {{11,12,13,14},{25,26,27,28},{39,40,41,42}};
    const int LW = 53, LB = 54;
    const int CIN[4] = {256, 320, 384, 448};

    auto F = [&](int i) { return (const float*)d_in[i]; };

    // workspace (floats, then bf16 weight tables; base is 16B aligned,
    // float region size 1,869,568 keeps the u16 region 16B aligned)
    float* xbuf = (float*)d_ws;                      // B*V*448
    float* Abuf = xbuf + (size_t)B_ * V_ * XS;       // B*V*V
    float* zbuf = Abuf + (size_t)NE_ADJ;             // B*V*64
    float* st   = zbuf + (size_t)B_ * V_ * 64;
    float *sum1 = st,        *sq1 = st + 256;
    float *sum2 = st + 512,  *sq2 = st + 640;
    float *sumg = st + 768,  *sqg = st + 832;        // sums region: 896 floats
    float *sc1  = st + 896,  *sh1 = st + 1152;
    float *sc2  = st + 1408, *sh2 = st + 1536;
    float *scg  = st + 1664, *shg = st + 1728;
    unsigned short* W1bf = (unsigned short*)(st + 1792);   // [256][cin] bf16
    unsigned short* W2bf = W1bf + (size_t)C1 * XS;         // [128][256] bf16

    float* outp  = (float*)d_out;                    // [32*5] then A_last [32*96*96]
    float* Alast = outp + B_ * 5;

    const float invA = 1.0f / (float)NE_ADJ;
    const float invG = 1.0f / (float)NE_G;
    const dim3 blk(256);

    k_init_x<<<B_ * V_, blk, 0, stream>>>(F(0), xbuf);

    for (int L = 0; L < 4; ++L) {
        const AdjIdx a = AJ[L];
        const int cin  = CIN[L];
        float* Aout    = (L == 3) ? Alast : Abuf;

        k_zero<<<4, blk, 0, stream>>>(st, 896);
        k_wprep<<<(cin * C1 + 255) / 256, blk, 0, stream>>>(F(a.w1), W1bf, cin, C1);
        k_wprep<<<(C1 * C2 + 255) / 256, blk, 0, stream>>>(F(a.w2), W2bf, C1, C2);

        k_adj<1><<<B_ * V_, blk, SMEM_ADJ, stream>>>(
            xbuf, cin, W1bf, F(a.b1), nullptr, nullptr,
            W2bf, F(a.b2), nullptr, nullptr, F(a.w3), F(a.b3),
            sum1, sq1, sum2, sq2, nullptr);
        k_finalize<<<1, blk, 0, stream>>>(sum1, sq1, F(a.g1), F(a.be1), sc1, sh1, C1, invA);

        k_adj<2><<<B_ * V_, blk, SMEM_ADJ, stream>>>(
            xbuf, cin, W1bf, F(a.b1), sc1, sh1,
            W2bf, F(a.b2), nullptr, nullptr, F(a.w3), F(a.b3),
            sum1, sq1, sum2, sq2, nullptr);
        k_finalize<<<1, blk, 0, stream>>>(sum2, sq2, F(a.g2), F(a.be2), sc2, sh2, C2, invA);

        k_adj<3><<<B_ * V_, blk, SMEM_ADJ, stream>>>(
            xbuf, cin, W1bf, F(a.b1), sc1, sh1,
            W2bf, F(a.b2), sc2, sh2, F(a.w3), F(a.b3),
            sum1, sq1, sum2, sq2, Aout);

        if (L < 3) {
            k_gconv_z<<<B_ * V_, blk, 0, stream>>>(
                xbuf, Abuf, F(GW[L][0]), F(GW[L][1]), zbuf, sumg, sqg, cin);
            k_finalize<<<1, blk, 0, stream>>>(sumg, sqg, F(GW[L][2]), F(GW[L][3]),
                                              scg, shg, 64, invG);
            k_gconv_apply<<<(B_ * V_ * 64) / 256, blk, 0, stream>>>(zbuf, scg, shg, xbuf, cin);
        }
    }

    k_last_out<<<B_, blk, 0, stream>>>(xbuf, Alast, F(LW), F(LB), outp);
}
